// Attention_46291157516784
// MI455X (gfx1250) — matrix operand order repaired and checked by reading, not compiled, not run on hardware
//
#include <hip/hip_runtime.h>

// MHA forward for B=2, T=2048, C=1024, H=16, D=64 on gfx1250 (wave32, WMMA).
// qkv bf16-WMMA GEMM -> flash attention (bf16 WMMA, fp32 softmax, TDM LDS staging)
// -> proj bf16-WMMA GEMM (async-load A staging).

typedef __attribute__((ext_vector_type(16))) __bf16        bf16x16;
typedef __attribute__((ext_vector_type(8)))  float         f32x8;
typedef __attribute__((ext_vector_type(4)))  unsigned int  u32x4;
typedef __attribute__((ext_vector_type(8)))  unsigned int  u32x8;
typedef __attribute__((ext_vector_type(4)))  float         f32x4;

#define DIMC   1024
#define NH     16
#define HD     64
#define SEQ    2048
#define BTROWS 4096            // B*T
#define LOG2E  1.4426950408889634f

__device__ __forceinline__ unsigned short f2bf(float f) {
  unsigned int u = __float_as_uint(f);
  u += 0x7FFFu + ((u >> 16) & 1u);          // round-to-nearest-even
  return (unsigned short)(u >> 16);
}

union Frag { u32x4 q[2]; bf16x16 v; };

// A-matrix 16x32 bf16 fragment (ISA 7.12.2): lane m=L&15, half=L>>4,
// elems 0..7 -> k = 8*half+0..7 ; elems 8..15 -> k = 16+8*half+0..7 (plus kOff).
__device__ __forceinline__ bf16x16 lds_frag_a(const unsigned short* row0, int stride, int kOff, int lane) {
  const int m = lane & 15, half = lane >> 4;
  const unsigned short* p = row0 + m * stride + kOff + 8 * half;
  Frag f;
  f.q[0] = *(const u32x4*)(p);
  f.q[1] = *(const u32x4*)(p + 16);
  return f.v;
}

// B-matrix 32x16 bf16 fragment: lane n=L&15 (column), elems 0..7 -> k = 8*half+0..7 ; elems 8..15 -> k = 16+8*half+0..7 (+kOff).
// LDS tile stored as [n rows][k cols] so this is two 16-byte reads.
__device__ __forceinline__ bf16x16 lds_frag_b(const unsigned short* row0, int stride, int kOff, int lane) {
  const int n = lane & 15, half = lane >> 4;
  const unsigned short* p = row0 + n * stride + kOff + 8 * half;
  Frag f;
  f.q[0] = *(const u32x4*)(p);
  f.q[1] = *(const u32x4*)(p + 16);
  return f.v;
}

__device__ __forceinline__ f32x8 wmma_bf16(bf16x16 a, bf16x16 b, f32x8 c) {
  return __builtin_amdgcn_wmma_f32_16x16x32_bf16(false, a, false, b, (short)0, c, false, false);
}

__device__ __forceinline__ unsigned int lds_off32(const void* p) {
  return (unsigned int)(unsigned long long)p;   // LDS aperture keeps offset in addr[31:0]
}

// CDNA5 async global->LDS copy of 64 contiguous bytes (ASYNCcnt-tracked).
__device__ __forceinline__ void async_copy64(unsigned int lds_off, const void* g) {
  asm volatile(
      "global_load_async_to_lds_b128 %0, %1, off\n\t"
      "global_load_async_to_lds_b128 %0, %1, off offset:16\n\t"
      "global_load_async_to_lds_b128 %0, %1, off offset:32\n\t"
      "global_load_async_to_lds_b128 %0, %1, off offset:48"
      :: "v"(lds_off), "v"(g) : "memory");
}
__device__ __forceinline__ void wait_async0() {
  asm volatile("s_wait_asynccnt 0" ::: "memory");
}

__device__ __forceinline__ unsigned int rfl(unsigned int x) {
  return (unsigned int)__builtin_amdgcn_readfirstlane((int)x);
}

// Tensor Data Mover: load a 64x64 tile of 2-byte elements (row stride
// `rowStrideElems` in global) into LDS at byte offset `ldsOff`, padding each
// 128B row by 16B (pad_interval=32 DWORDs, pad_amount=4 DWORDs) -> LDS row
// stride 72 halfwords, matching the WMMA fragment loaders. D# layout per
// CDNA5 ISA (2 SGPR groups, VADDR2/3 = NULL -> 2D).
__device__ __forceinline__ void tdm_load_tile_2d(unsigned int ldsOff, const void* g,
                                                 unsigned int rowStrideElems) {
  const unsigned long long ga = (unsigned long long)g;
  u32x4 g0;
  g0[0] = rfl(1u);                                       // count=1, user descriptor
  g0[1] = rfl(ldsOff);                                   // lds_addr (bytes)
  g0[2] = rfl((unsigned int)ga);                         // global_addr[31:0]
  g0[3] = rfl(((unsigned int)(ga >> 32) & 0x1FFFFFFu)    // global_addr[56:32]
              | (2u << 30));                             // type=2 ("image")
  u32x8 g1;
  g1[0] = rfl((1u << 16)                                 // data_size = 2 bytes
            | (1u << 20)                                 // pad_enable
            | (4u << 22)                                 // pad_interval: 32 DWORDs
            | (3u << 25));                               // pad_amount: 4 DWORDs (16B)
  g1[1] = rfl(64u << 16);                                // tensor_dim0 = 64 (lo16)
  g1[2] = rfl(2048u << 16);                              // dim0 hi=0 ; tensor_dim1 lo = 2048
  g1[3] = rfl(64u << 16);                                // dim1 hi=0 ; tile_dim0 = 64
  g1[4] = rfl(64u);                                      // tile_dim1 = 64 ; tile_dim2 = 0
  g1[5] = rfl(rowStrideElems);                           // tensor_dim0_stride lo32
  g1[6] = rfl(0u);                                       // stride hi ; dim1_stride lo
  g1[7] = rfl(0u);                                       // dim1_stride hi (unused, 2D)
  asm volatile("tensor_load_to_lds %0, %1" :: "s"(g0), "s"(g1) : "memory");
}

// ---------------------------------------------------------------------------
// Kernel 1: qkv = x @ w_qkv + b_qkv ; scatter into Q(B,H,T,D)*scale, K(B,H,T,D),
// V(B,H,D,T), all bf16. Block = 256 threads (8 waves), tile 128x128, K-step 64.
// ---------------------------------------------------------------------------
__global__ __launch_bounds__(256) void qkv_gemm_kernel(
    const float* __restrict__ x, const float* __restrict__ w, const float* __restrict__ bias,
    unsigned short* __restrict__ Qs, unsigned short* __restrict__ Kb, unsigned short* __restrict__ Vt) {
  __shared__ unsigned short As[128 * 72];   // [m][k0..63], stride 72 halves (144B rows)
  __shared__ unsigned short Bs[128 * 72];   // [n][k] (transposed weight tile)
  const int tid = threadIdx.x, lane = tid & 31, wv = tid >> 5;
  const int nb = blockIdx.x * 128, mb = blockIdx.y * 128;

  f32x8 acc[8];
  for (int i = 0; i < 8; ++i)
    for (int j = 0; j < 8; ++j) acc[i][j] = 0.f;

  const int srow = tid >> 1, scol = (tid & 1) * 32;

  for (int kk = 0; kk < DIMC; kk += 64) {
    __syncthreads();
    { // stage A: x fp32 -> bf16, 32 contiguous elements per thread
      const float* src = x + (size_t)(mb + srow) * DIMC + kk + scol;
      unsigned short* dst = As + srow * 72 + scol;
      #pragma unroll
      for (int v = 0; v < 8; ++v) {
        f32x4 t = ((const f32x4*)src)[v];
        dst[4 * v + 0] = f2bf(t[0]);
        dst[4 * v + 1] = f2bf(t[1]);
        dst[4 * v + 2] = f2bf(t[2]);
        dst[4 * v + 3] = f2bf(t[3]);
      }
    }
    { // stage B transposed: Bs[n][k] = w[kk+k][nb+n]
      const float* src = w + (size_t)(kk + scol) * 3072 + nb + srow;
      unsigned short* dst = Bs + srow * 72 + scol;
      #pragma unroll
      for (int i = 0; i < 32; ++i) dst[i] = f2bf(src[(size_t)i * 3072]);
    }
    __syncthreads();

    // Two k-steps of 32; pipeline B-fragment loads ahead of the WMMA pair.
    bf16x16 af0 = lds_frag_a(As + wv * 16 * 72, 72, 0, lane);
    bf16x16 af1 = lds_frag_a(As + wv * 16 * 72, 72, 32, lane);
    bf16x16 b0 = lds_frag_b(Bs, 72, 0, lane);
    bf16x16 b1 = lds_frag_b(Bs, 72, 32, lane);
    #pragma unroll
    for (int nt = 0; nt < 8; ++nt) {
      bf16x16 n0 = b0, n1 = b1;
      if (nt < 7) {
        n0 = lds_frag_b(Bs + (nt + 1) * 16 * 72, 72, 0, lane);
        n1 = lds_frag_b(Bs + (nt + 1) * 16 * 72, 72, 32, lane);
      }
      acc[nt] = wmma_bf16(af0, b0, acc[nt]);
      acc[nt] = wmma_bf16(af1, b1, acc[nt]);
      b0 = n0; b1 = n1;
    }
  }

  // Epilogue: C layout = vgpr r -> row r+8*half, lane&15 -> col.
  const int nlo = lane & 15, half = lane >> 4;
  const float qscale = 0.125f * LOG2E;                 // D^-0.5 * log2(e), folded into Q
  for (int nt = 0; nt < 8; ++nt) {
    const int n = nb + nt * 16 + nlo;
    const float bv = bias[n];
    const int sec = n >> 10, cc = n & 1023, h = cc >> 6, d = cc & 63;
    for (int r = 0; r < 8; ++r) {
      const int trow = mb + wv * 16 + r + 8 * half;
      const int b = trow >> 11, t = trow & 2047;
      const size_t bh = (size_t)(b * NH + h);
      const float val = acc[nt][r] + bv;
      if (sec == 0)      Qs[(bh * SEQ + t) * HD + d] = f2bf(val * qscale);
      else if (sec == 1) Kb[(bh * SEQ + t) * HD + d] = f2bf(val);
      else               Vt[(bh * HD + d) * SEQ + t] = f2bf(val);
    }
  }
}

// ---------------------------------------------------------------------------
// Kernel 2: flash attention. Block = 128 threads (4 waves), 64 queries per block,
// one (b,h) per 32 blocks. 64-key K/V tiles staged by the Tensor Data Mover.
// ---------------------------------------------------------------------------
__global__ __launch_bounds__(128) void flash_attn_kernel(
    const unsigned short* __restrict__ Qs, const unsigned short* __restrict__ Kb,
    const unsigned short* __restrict__ Vt, unsigned short* __restrict__ Ob) {
  __shared__ unsigned short Ks[64 * 72];        // [key][d]   (TDM-padded rows)
  __shared__ unsigned short Vs[64 * 72];        // [d][key]   (V pre-transposed in global)
  __shared__ unsigned short Ps[4 * 16 * 72];    // per-wave P tile [m][k]

  const int tid = threadIdx.x, lane = tid & 31, wv = tid >> 5;
  const int qt = blockIdx.x & 31;               // query tile (T/64 = 32)
  const int bh = blockIdx.x >> 5;               // b*NH + h
  const int h = bh & (NH - 1), b = bh >> 4;

  const unsigned short* qptr = Qs + ((size_t)bh * SEQ + qt * 64 + wv * 16) * HD;
  const unsigned short* kptr = Kb + (size_t)bh * SEQ * HD;
  const unsigned short* vptr = Vt + (size_t)bh * HD * SEQ;

  const int m = lane & 15, half = lane >> 4, nlo = lane & 15;

  // Q A-fragments (2 k-steps of 32 over d=64), loaded once from global.
  bf16x16 qf[2];
  for (int ks = 0; ks < 2; ++ks) {
    const unsigned short* p = qptr + m * HD + ks * 32 + 8 * half;
    Frag f;
    f.q[0] = *(const u32x4*)(p);
    f.q[1] = *(const u32x4*)(p + 16);
    qf[ks] = f.v;
  }

  float mrow[8], lrow[8];
  f32x8 oacc[4];
  for (int r = 0; r < 8; ++r) { mrow[r] = -3.0e38f; lrow[r] = 0.f; }
  for (int nt = 0; nt < 4; ++nt)
    for (int j = 0; j < 8; ++j) oacc[nt][j] = 0.f;

  const unsigned int ksBase = lds_off32(Ks);
  const unsigned int vsBase = lds_off32(Vs);
  const int srow = tid >> 1;

  for (int j0 = 0; j0 < SEQ; j0 += 64) {
    if (j0 + 64 < SEQ) {  // warm L2 ahead of the next TDM tiles
      __builtin_prefetch(kptr + (size_t)(j0 + 64 + srow) * HD, 0, 1);
      __builtin_prefetch(vptr + (size_t)srow * SEQ + j0 + 64, 0, 1);
    }
    __syncthreads();
    // TDM-stage K tile [key][d] (row stride 64) and V tile [d][key] (row stride 2048).
    if (wv == 0)      tdm_load_tile_2d(ksBase, kptr + (size_t)j0 * HD, 64u);
    else if (wv == 1) tdm_load_tile_2d(vsBase, vptr + j0, 2048u);
    __builtin_amdgcn_s_wait_tensorcnt(0);
    __syncthreads();

    // S = Q @ K^T  (4 n-tiles of 16 keys; k-dim = d split 2x32), pipelined frags.
    f32x8 sacc[4];
    bf16x16 kb0 = lds_frag_b(Ks, 72, 0, lane);
    bf16x16 kb1 = lds_frag_b(Ks, 72, 32, lane);
    #pragma unroll
    for (int nt = 0; nt < 4; ++nt) {
      bf16x16 n0 = kb0, n1 = kb1;
      if (nt < 3) {
        n0 = lds_frag_b(Ks + (nt + 1) * 16 * 72, 72, 0, lane);
        n1 = lds_frag_b(Ks + (nt + 1) * 16 * 72, 72, 32, lane);
      }
      f32x8 z;
      for (int j = 0; j < 8; ++j) z[j] = 0.f;
      z = wmma_bf16(qf[0], kb0, z);
      z = wmma_bf16(qf[1], kb1, z);
      sacc[nt] = z;
      kb0 = n0; kb1 = n1;
    }

    // Online softmax (scores already in log2 domain: Q pre-scaled by scale*log2e).
    float mnew[8], fscale[8];
    for (int r = 0; r < 8; ++r) {
      float tm = fmaxf(fmaxf(sacc[0][r], sacc[1][r]), fmaxf(sacc[2][r], sacc[3][r]));
      tm = fmaxf(tm, __shfl_xor(tm, 1, 32));
      tm = fmaxf(tm, __shfl_xor(tm, 2, 32));
      tm = fmaxf(tm, __shfl_xor(tm, 4, 32));
      tm = fmaxf(tm, __shfl_xor(tm, 8, 32));     // row max within 16-lane group
      const float mn = fmaxf(mrow[r], tm);
      mnew[r] = mn;
      fscale[r] = exp2f(mrow[r] - mn);
      mrow[r] = mn;
    }
    for (int r = 0; r < 8; ++r) {
      float s = 0.f;
      for (int nt = 0; nt < 4; ++nt) {
        const float p = exp2f(sacc[nt][r] - mnew[r]);
        sacc[nt][r] = p;
        s += p;
      }
      s += __shfl_xor(s, 1, 32);
      s += __shfl_xor(s, 2, 32);
      s += __shfl_xor(s, 4, 32);
      s += __shfl_xor(s, 8, 32);
      lrow[r] = lrow[r] * fscale[r] + s;
      for (int nt = 0; nt < 4; ++nt) oacc[nt][r] *= fscale[r];
    }

    // P -> LDS (per-wave private region; LDS in-order per wave, no barrier needed)
    unsigned short* pbase = Ps + wv * 16 * 72;
    for (int nt = 0; nt < 4; ++nt)
      for (int r = 0; r < 8; ++r)
        pbase[(r + 8 * half) * 72 + nt * 16 + nlo] = f2bf(sacc[nt][r]);

    // O += P @ V  (k-dim = 64 keys split 2x32; n-dim = d, 4 tiles of 16)
    #pragma unroll
    for (int ks = 0; ks < 2; ++ks) {
      bf16x16 pf = lds_frag_a(pbase, 72, ks * 32, lane);
      bf16x16 v0 = lds_frag_b(Vs, 72, ks * 32, lane);
      #pragma unroll
      for (int nt = 0; nt < 4; ++nt) {
        bf16x16 nv = v0;
        if (nt < 3) nv = lds_frag_b(Vs + (nt + 1) * 16 * 72, 72, ks * 32, lane);
        oacc[nt] = wmma_bf16(pf, v0, oacc[nt]);
        v0 = nv;
      }
    }
  }

  // Normalize and store to Ob laid out (B, T, H, D) == (4096, 1024) bf16.
  for (int r = 0; r < 8; ++r) {
    const float inv = 1.0f / lrow[r];
    const int trow = b * SEQ + qt * 64 + wv * 16 + r + 8 * half;
    const size_t rowoff = (size_t)trow * DIMC + h * HD;
    for (int nt = 0; nt < 4; ++nt)
      Ob[rowoff + nt * 16 + nlo] = f2bf(oacc[nt][r] * inv);
  }
}

// ---------------------------------------------------------------------------
// Kernel 3: out = Ob @ w_proj + b_proj (fp32 out). A tile staged via async loads.
// ---------------------------------------------------------------------------
__global__ __launch_bounds__(256) void proj_gemm_kernel(
    const unsigned short* __restrict__ Ob, const float* __restrict__ w,
    const float* __restrict__ bias, float* __restrict__ out) {
  __shared__ unsigned short As[128 * 72];
  __shared__ unsigned short Bs[128 * 72];
  const int tid = threadIdx.x, lane = tid & 31, wv = tid >> 5;
  const int nb = blockIdx.x * 128, mb = blockIdx.y * 128;

  f32x8 acc[8];
  for (int i = 0; i < 8; ++i)
    for (int j = 0; j < 8; ++j) acc[i][j] = 0.f;

  const int srow = tid >> 1, scol = (tid & 1) * 32;
  const unsigned int asLds = lds_off32(As + srow * 72 + scol);

  for (int kk = 0; kk < DIMC; kk += 64) {
    __syncthreads();
    // stage A (already bf16): async 64B copy per thread
    async_copy64(asLds, Ob + (size_t)(mb + srow) * DIMC + kk + scol);
    { // stage B transposed: Bs[n][k] = w[kk+k][nb+n]
      const float* src = w + (size_t)(kk + scol) * DIMC + nb + srow;
      unsigned short* dst = Bs + srow * 72 + scol;
      #pragma unroll
      for (int i = 0; i < 32; ++i) dst[i] = f2bf(src[(size_t)i * DIMC]);
    }
    wait_async0();
    __syncthreads();

    bf16x16 af0 = lds_frag_a(As + wv * 16 * 72, 72, 0, lane);
    bf16x16 af1 = lds_frag_a(As + wv * 16 * 72, 72, 32, lane);
    bf16x16 b0 = lds_frag_b(Bs, 72, 0, lane);
    bf16x16 b1 = lds_frag_b(Bs, 72, 32, lane);
    #pragma unroll
    for (int nt = 0; nt < 8; ++nt) {
      bf16x16 n0 = b0, n1 = b1;
      if (nt < 7) {
        n0 = lds_frag_b(Bs + (nt + 1) * 16 * 72, 72, 0, lane);
        n1 = lds_frag_b(Bs + (nt + 1) * 16 * 72, 72, 32, lane);
      }
      acc[nt] = wmma_bf16(af0, b0, acc[nt]);
      acc[nt] = wmma_bf16(af1, b1, acc[nt]);
      b0 = n0; b1 = n1;
    }
  }

  const int nlo = lane & 15, half = lane >> 4;
  for (int nt = 0; nt < 8; ++nt) {
    const int n = nb + nt * 16 + nlo;
    const float bv = bias[n];
    for (int r = 0; r < 8; ++r) {
      const int trow = mb + wv * 16 + r + 8 * half;
      out[(size_t)trow * DIMC + n] = acc[nt][r] + bv;
    }
  }
}

extern "C" void kernel_launch(void* const* d_in, const int* in_sizes, int n_in,
                              void* d_out, int out_size, void* d_ws, size_t ws_size,
                              hipStream_t stream) {
  (void)in_sizes; (void)n_in; (void)out_size; (void)ws_size;
  const float* x      = (const float*)d_in[0];
  const float* w_qkv  = (const float*)d_in[1];
  const float* b_qkv  = (const float*)d_in[2];
  const float* w_proj = (const float*)d_in[3];
  const float* b_proj = (const float*)d_in[4];
  float* out = (float*)d_out;

  // Workspace: Q, K, V (each B*H*T*D bf16) + attention output (B*T*C bf16) = 32 MB.
  const size_t elems = (size_t)2 * NH * SEQ * HD;
  unsigned short* Qs = (unsigned short*)d_ws;
  unsigned short* Kb = Qs + elems;
  unsigned short* Vt = Kb + elems;
  unsigned short* Ob = Vt + elems;

  qkv_gemm_kernel<<<dim3(3072 / 128, BTROWS / 128), 256, 0, stream>>>(x, w_qkv, b_qkv, Qs, Kb, Vt);
  flash_attn_kernel<<<dim3((2 * NH) * (SEQ / 64)), 128, 0, stream>>>(Qs, Kb, Vt, Ob);
  proj_gemm_kernel<<<dim3(DIMC / 128, BTROWS / 128), 256, 0, stream>>>(Ob, w_proj, b_proj, out);
}
